// GATv2_raw_att_12_68968584839887
// MI455X (gfx1250) — matrix operand order repaired and checked by reading, not compiled, not run on hardware
//
#include <hip/hip_runtime.h>
#include <hip/hip_bf16.h>
#include <cstdint>
#include <cstddef>

typedef __attribute__((ext_vector_type(16))) __bf16 v16bf;
typedef __attribute__((ext_vector_type(8)))  float  v8f;

#define NNODES 12288
#define NEDGES 98304
#define NE2    110592
#define NGRAPH 48
#define NPER   256
#define KKEEP  128
#define EDIM   6

// ---------------- helpers ----------------
__device__ __forceinline__ unsigned int fenc(float f) {
  unsigned int u = __float_as_uint(f);
  return (u & 0x80000000u) ? ~u : (u | 0x80000000u);
}
__device__ __forceinline__ float fdec(unsigned int u) {
  return __uint_as_float((u & 0x80000000u) ? (u & 0x7fffffffu) : ~u);
}
__device__ __forceinline__ unsigned short f2bf(float f) {
  unsigned int u = __float_as_uint(f);
  return (unsigned short)((u + 0x7FFFu + ((u >> 16) & 1u)) >> 16);
}

// f32 -> bf16 (RNE)
__global__ __launch_bounds__(256) void cvt_bf16(const float* __restrict__ in,
                                                unsigned short* __restrict__ out, int n) {
  int i = blockIdx.x * 256 + threadIdx.x;
  if (i < n) out[i] = f2bf(in[i]);
}
// f32 (rows of fin) -> bf16 rows padded to fout (dest pre-zeroed)
__global__ __launch_bounds__(256) void cvt_bf16_pad(const float* __restrict__ in,
    unsigned short* __restrict__ out, int n, int fin, int fout) {
  int i = blockIdx.x * 256 + threadIdx.x;
  if (i < n) {
    int r = i / fin, c = i - r * fin;
    out[(size_t)r * fout + c] = f2bf(in[i]);
  }
}

// ---------------- WMMA GEMM: C[M,N] = A[M,K](bf16) @ B[K,N](bf16) + bias ----------------
// Requirements: K % 32 == 0, N % 64 == 0. M arbitrary. Block tile 128x64, 8 waves,
// each wave computes a 2x2 grid of 16x16 WMMA tiles (4 wmma / K-step).
__global__ __launch_bounds__(256) void gemm_bf16_wmma(
    const unsigned short* __restrict__ A, const unsigned short* __restrict__ Bw,
    const float* __restrict__ bias, float* __restrict__ Cm, int M, int N, int K) {
  __shared__ unsigned short As[128][40];  // [m][k], stride 40 (b128-aligned, conflict-free)
  __shared__ unsigned short Bs[64][34];   // transposed [n][k], stride 34 (17-word reads)
  const int tid  = threadIdx.x;
  const int lane = tid & 31;
  const int wave = tid >> 5;
  const int m0 = blockIdx.y * 128;
  const int n0 = blockIdx.x * 64;
  const int wr = wave & 3;          // wave tile-row pair: rows 2*wr, 2*wr+1
  const int wc = wave >> 2;         // wave tile-col pair: cols 2*wc, 2*wc+1
  const int half = lane >> 4;
  const int l15  = lane & 15;

  // loop-invariant load coordinates
  const int arow_ld = tid >> 1;             // 0..127
  const int akoff   = (tid & 1) * 16;       // 0 or 16
  const bool arow_ok = (m0 + arow_ld) < M;
  const unsigned short* aptr = A + (size_t)(m0 + arow_ld) * K + akoff;
  const int bkk = tid >> 3;                 // 0..31
  const int bnn = (tid & 7) * 8;            // 0..56
  const unsigned short* bptr = Bw + (size_t)bkk * N + n0 + bnn;

  v8f c00 = {0.f,0.f,0.f,0.f,0.f,0.f,0.f,0.f};
  v8f c01 = c00, c10 = c00, c11 = c00;

  for (int k0 = 0; k0 < K; k0 += 32) {
    __syncthreads();
    // ---- A tile 128x32: two b128 loads per thread (row guard is loop-invariant) ----
    uint4 qa0 = {0u,0u,0u,0u}, qa1 = {0u,0u,0u,0u};
    if (arow_ok) {
      qa0 = *(const uint4*)(aptr + k0);
      qa1 = *(const uint4*)(aptr + k0 + 8);
    }
    *(uint4*)&As[arow_ld][akoff]     = qa0;
    *(uint4*)&As[arow_ld][akoff + 8] = qa1;
    // ---- B tile 32x64: one b128 load, transposed b16 stores ----
    union { uint4 q; unsigned short u[8]; } qb;
    qb.q = *(const uint4*)(bptr + (size_t)k0 * N);
#pragma unroll
    for (int j = 0; j < 8; ++j) Bs[bnn + j][bkk] = qb.u[j];

    if (k0 + 32 < K) {  // CDNA5 global_prefetch_b8 of next tiles
      __builtin_prefetch(aptr + k0 + 32, 0, 1);
      __builtin_prefetch(bptr + (size_t)(k0 + 32) * N, 0, 1);
    }
    __syncthreads();

    // ---- fragments ----
    union { v16bf v; unsigned int w[8]; } a0, a1, b0, b1;
    const int ar0 = (2 * wr) * 16 + l15;
    const int ar1 = ar0 + 16;
#pragma unroll
    for (int v = 0; v < 8; ++v) {           // ISA 16-bit A 16x32 per-lane K striping
      int kk = ((v >> 2) << 4) + half * 8 + ((v & 3) << 1);
      a0.w[v] = *(const unsigned int*)&As[ar0][kk];
      a1.w[v] = *(const unsigned int*)&As[ar1][kk];
    }
    const int bc0 = (2 * wc) * 16 + l15;
#pragma unroll
    for (int v = 0; v < 8; ++v) {           // B: lane=col, K striping same as A
      int kk = ((v >> 2) << 4) + half * 8 + ((v & 3) << 1);
      b0.w[v] = *(const unsigned int*)&Bs[bc0][kk];
      b1.w[v] = *(const unsigned int*)&Bs[bc0 + 16][kk];
    }
    c00 = __builtin_amdgcn_wmma_f32_16x16x32_bf16(false, a0.v, false, b0.v, (short)0, c00, false, false);
    c01 = __builtin_amdgcn_wmma_f32_16x16x32_bf16(false, a0.v, false, b1.v, (short)0, c01, false, false);
    c10 = __builtin_amdgcn_wmma_f32_16x16x32_bf16(false, a1.v, false, b0.v, (short)0, c10, false, false);
    c11 = __builtin_amdgcn_wmma_f32_16x16x32_bf16(false, a1.v, false, b1.v, (short)0, c11, false, false);
  }

  const int crow0 = m0 + (2 * wr) * 16 + half * 8;
  const int crow1 = crow0 + 16;
  const int ccol0 = n0 + (2 * wc) * 16 + l15;
  const int ccol1 = ccol0 + 16;
  float bb0 = bias ? bias[ccol0] : 0.f;
  float bb1 = bias ? bias[ccol1] : 0.f;
#pragma unroll
  for (int r = 0; r < 8; ++r) {
    int g0r = crow0 + r, g1r = crow1 + r;
    if (g0r < M) {
      Cm[(size_t)g0r * N + ccol0] = c00[r] + bb0;
      Cm[(size_t)g0r * N + ccol1] = c01[r] + bb1;
    }
    if (g1r < M) {
      Cm[(size_t)g1r * N + ccol0] = c10[r] + bb0;
      Cm[(size_t)g1r * N + ccol1] = c11[r] + bb1;
    }
  }
}

// ---------------- self-loop attr (fill_value='mean') ----------------
__global__ __launch_bounds__(256) void edge_count_sum(const int* __restrict__ dst,
    const float* __restrict__ ef, float* __restrict__ cnt, float* __restrict__ lsum, int E) {
  int e = blockIdx.x * 256 + threadIdx.x;
  if (e < E) {
    int d = dst[e];
    atomicAdd(&cnt[d], 1.f);
#pragma unroll
    for (int k = 0; k < EDIM; ++k) atomicAdd(&lsum[(size_t)d * EDIM + k], ef[(size_t)e * EDIM + k]);
  }
}
__global__ __launch_bounds__(256) void loop_attr_fin(float* __restrict__ lattr,
                                                     const float* __restrict__ cnt, int n) {
  int i = blockIdx.x * 256 + threadIdx.x;
  if (i < n) lattr[i] /= fmaxf(cnt[i / EDIM], 1.f);
}

// ---------------- fused GATv2 attention logits (H=4) ----------------
__global__ __launch_bounds__(256) void edge_alpha_kernel(
    const float* __restrict__ XL, const float* __restrict__ XR,
    const float* __restrict__ We, const float* __restrict__ EF,
    const float* __restrict__ lattr, const float* __restrict__ att,
    const int* __restrict__ srcA, const int* __restrict__ dstA,
    float* __restrict__ alpha, int HC, int logC, int E, int E2) {
  int e = blockIdx.x;
  if (e >= E2) return;
  int s, d; const float* ea;
  if (e < E) { s = srcA[e]; d = dstA[e]; ea = EF + (size_t)e * EDIM; }
  else       { s = d = e - E;            ea = lattr + (size_t)(e - E) * EDIM; }
  float e0 = ea[0], e1 = ea[1], e2 = ea[2], e3 = ea[3], e4 = ea[4], e5 = ea[5];
  const float* xl = XL + (size_t)s * HC;
  const float* xr = XR + (size_t)d * HC;
  float acc[4] = {0.f, 0.f, 0.f, 0.f};
  for (int c = threadIdx.x; c < HC; c += 256) {
    float z = xl[c] + xr[c] + e0 * We[c] + e1 * We[HC + c] + e2 * We[2 * HC + c]
            + e3 * We[3 * HC + c] + e4 * We[4 * HC + c] + e5 * We[5 * HC + c];
    z = z > 0.f ? z : 0.2f * z;
    acc[c >> logC] += z * att[c];
  }
  __shared__ float red[4][256];
#pragma unroll
  for (int h = 0; h < 4; ++h) red[h][threadIdx.x] = acc[h];
  __syncthreads();
  for (int off = 128; off; off >>= 1) {
    if ((int)threadIdx.x < off)
#pragma unroll
      for (int h = 0; h < 4; ++h) red[h][threadIdx.x] += red[h][threadIdx.x + off];
    __syncthreads();
  }
  if (threadIdx.x < 4) alpha[(size_t)e * 4 + threadIdx.x] = red[threadIdx.x][0];
}

// ---------------- segment softmax pieces ----------------
__global__ __launch_bounds__(256) void seg_max(const float* __restrict__ alpha,
    const int* __restrict__ dstA, unsigned int* __restrict__ menc, int E, int E2, int H) {
  int t = blockIdx.x * 256 + threadIdx.x;
  if (t >= E2 * H) return;
  int e = t / H, h = t - e * H;
  int d = (e < E) ? dstA[e] : (e - E);
  atomicMax(&menc[(size_t)d * H + h], fenc(alpha[t]));
}
__global__ __launch_bounds__(256) void seg_expsum(float* __restrict__ alpha,
    const int* __restrict__ dstA, const unsigned int* __restrict__ menc,
    float* __restrict__ den, int E, int E2, int H) {
  int t = blockIdx.x * 256 + threadIdx.x;
  if (t >= E2 * H) return;
  int e = t / H, h = t - e * H;
  int d = (e < E) ? dstA[e] : (e - E);
  float a = __expf(alpha[t] - fdec(menc[(size_t)d * H + h]));
  alpha[t] = a;
  atomicAdd(&den[(size_t)d * H + h], a);
}

// ---------------- weighted aggregation (H=4) ----------------
__global__ __launch_bounds__(256) void aggregate(
    const float* __restrict__ XL, const float* __restrict__ alpha, const float* __restrict__ den,
    const int* __restrict__ srcA, const int* __restrict__ dstA, float* __restrict__ out,
    int HC, int logC, int E, int E2) {
  int e = blockIdx.x;
  if (e >= E2) return;
  int s, d;
  if (e < E) { s = srcA[e]; d = dstA[e]; } else { s = d = e - E; }
  float w[4];
#pragma unroll
  for (int h = 0; h < 4; ++h) w[h] = alpha[(size_t)e * 4 + h] / (den[(size_t)d * 4 + h] + 1e-16f);
  const float* xl = XL + (size_t)s * HC;
  float* o = out + (size_t)d * HC;
  for (int c = threadIdx.x; c < HC; c += 256) atomicAdd(&o[c], xl[c] * w[c >> logC]);
}

// ---------------- batchnorm (+ ELU) ----------------
__global__ __launch_bounds__(256) void bn_stats(const float* __restrict__ x,
    float* __restrict__ sums, int Nrows, int F, int rowsPer) {
  int f = blockIdx.x * 256 + threadIdx.x;
  if (f >= F) return;
  int r0 = blockIdx.y * rowsPer;
  int r1 = min(r0 + rowsPer, Nrows);
  float s = 0.f, q = 0.f;
  for (int r = r0; r < r1; ++r) { float v = x[(size_t)r * F + f]; s += v; q += v * v; }
  atomicAdd(&sums[f], s);
  atomicAdd(&sums[F + f], q);
}
__global__ __launch_bounds__(256) void bn_elu(const float* __restrict__ x,
    const float* __restrict__ sums, const float* __restrict__ g, const float* __restrict__ b,
    float* __restrict__ y, int Nrows, int F) {
  size_t i = (size_t)blockIdx.x * 256 + threadIdx.x;
  if (i >= (size_t)Nrows * F) return;
  int f = (int)(i % F);
  float mu  = sums[f] / Nrows;
  float var = sums[F + f] / Nrows - mu * mu;
  float v = g[f] * (x[i] - mu) * rsqrtf(var + 1e-5f) + b[f];
  y[i] = v > 0.f ? v : (__expf(v) - 1.f);
}

// ---------------- score head (GATv2 H=1,C=1) ----------------
__global__ __launch_bounds__(256) void score_gemv(const float* __restrict__ h,
    const float* __restrict__ Wl, const float* __restrict__ Wr,
    const float* __restrict__ blp, const float* __restrict__ brp,
    float* __restrict__ xls, float* __restrict__ xrs, int F) {
  __shared__ float rl[256], rr[256];
  int n = blockIdx.x;
  float al = 0.f, ar = 0.f;
  for (int c = threadIdx.x; c < F; c += 256) {
    float v = h[(size_t)n * F + c];
    al += v * Wl[c]; ar += v * Wr[c];
  }
  rl[threadIdx.x] = al; rr[threadIdx.x] = ar;
  __syncthreads();
  for (int off = 128; off; off >>= 1) {
    if ((int)threadIdx.x < off) { rl[threadIdx.x] += rl[threadIdx.x + off]; rr[threadIdx.x] += rr[threadIdx.x + off]; }
    __syncthreads();
  }
  if (threadIdx.x == 0) { xls[n] = rl[0] + blp[0]; xrs[n] = rr[0] + brp[0]; }
}
__global__ __launch_bounds__(256) void score_edge(const float* __restrict__ xls,
    const float* __restrict__ xrs, const float* __restrict__ attp,
    const int* __restrict__ srcA, const int* __restrict__ dstA,
    float* __restrict__ alpha, int E, int E2) {
  int e = blockIdx.x * 256 + threadIdx.x;
  if (e >= E2) return;
  int s, d;
  if (e < E) { s = srcA[e]; d = dstA[e]; } else { s = d = e - E; }
  float z = xls[s] + xrs[d];
  z = z > 0.f ? z : 0.2f * z;
  alpha[e] = z * attp[0];
}
__global__ __launch_bounds__(256) void score_agg(const float* __restrict__ xls,
    const float* __restrict__ alpha, const float* __restrict__ den,
    const int* __restrict__ srcA, const int* __restrict__ dstA,
    float* __restrict__ outsc, int E, int E2) {
  int e = blockIdx.x * 256 + threadIdx.x;
  if (e >= E2) return;
  int s, d;
  if (e < E) { s = srcA[e]; d = dstA[e]; } else { s = d = e - E; }
  atomicAdd(&outsc[d], xls[s] * (alpha[e] / (den[d] + 1e-16f)));
}
__global__ __launch_bounds__(256) void score_tanh(const float* __restrict__ outsc,
    const float* __restrict__ bp, float* __restrict__ s, int n) {
  int i = blockIdx.x * 256 + threadIdx.x;
  if (i < n) s[i] = tanhf(outsc[i] + bp[0]);
}

// ---------------- SAGPool top-k + weighted mean pool ----------------
__global__ __launch_bounds__(NPER) void topk_mask(const float* __restrict__ s,
                                                  float* __restrict__ keepval) {
  __shared__ float sv[NPER];
  int b = blockIdx.x, i = threadIdx.x;
  float mine = s[b * NPER + i];
  sv[i] = mine;
  __syncthreads();
  int rank = 0;
  for (int j = 0; j < NPER; ++j) {
    float o = sv[j];
    rank += (o > mine) || (o == mine && j < i);
  }
  keepval[b * NPER + i] = (rank < KKEEP) ? mine : 0.f;
}
__global__ __launch_bounds__(256) void pool_mean(const float* __restrict__ h,
    const float* __restrict__ keepval, float* __restrict__ g, int F) {
  int f = blockIdx.x * 256 + threadIdx.x;
  int b = blockIdx.y;
  if (f >= F) return;
  float acc = 0.f;
  for (int i = 0; i < NPER; ++i)
    acc += keepval[b * NPER + i] * h[(size_t)(b * NPER + i) * F + f];
  g[(size_t)b * F + f] = acc * (1.f / KKEEP);
}

// ---------------- fc4 + log_softmax ----------------
__global__ __launch_bounds__(64) void fc4_logsoftmax(const float* __restrict__ g3,
    const float* __restrict__ W4, const float* __restrict__ b4, float* __restrict__ out) {
  __shared__ float r0[64], r1[64];
  int b = blockIdx.x, t = threadIdx.x;
  float v = g3[b * 64 + t];
  r0[t] = v * W4[t * 2 + 0];
  r1[t] = v * W4[t * 2 + 1];
  __syncthreads();
  for (int off = 32; off; off >>= 1) {
    if (t < off) { r0[t] += r0[t + off]; r1[t] += r1[t + off]; }
    __syncthreads();
  }
  if (t == 0) {
    float o0 = r0[0] + b4[0], o1 = r1[0] + b4[1];
    float m = fmaxf(o0, o1);
    float lse = m + __logf(__expf(o0 - m) + __expf(o1 - m));
    out[b * 2 + 0] = o0 - lse;
    out[b * 2 + 1] = o1 - lse;
  }
}

// ---------------- host orchestration ----------------
extern "C" void kernel_launch(void* const* d_in, const int* in_sizes, int n_in,
                              void* d_out, int out_size, void* d_ws, size_t ws_size,
                              hipStream_t stream) {
  (void)in_sizes; (void)n_in; (void)out_size; (void)ws_size;
  const float* x    = (const float*)d_in[0];
  const int*   eidx = (const int*)d_in[1];
  const int*   srcA = eidx;
  const int*   dstA = eidx + NEDGES;
  const float* ef   = (const float*)d_in[2];
  auto P = [&](int i) { return (const float*)d_in[i]; };

  // workspace carve
  char* wp = (char*)d_ws;
  auto alloc = [&](size_t bytes) -> void* {
    void* p = (void*)wp;
    wp += (bytes + 255) & ~(size_t)255;
    return p;
  };
  float*          h      = (float*)alloc((size_t)NNODES * 2048 * 4);
  unsigned short* hb     = (unsigned short*)alloc((size_t)NNODES * 2048 * 2);
  float*          XL     = (float*)alloc((size_t)NNODES * 2048 * 4);
  float*          XR     = (float*)alloc((size_t)NNODES * 2048 * 4);
  float*          alpha  = (float*)alloc((size_t)NE2 * 4 * 4);
  unsigned int*   menc   = (unsigned int*)alloc((size_t)NNODES * 4 * 4);
  float*          den    = (float*)alloc((size_t)NNODES * 4 * 4);
  float*          cnt    = (float*)alloc((size_t)NNODES * 4);
  float*          lattr  = (float*)alloc((size_t)NNODES * EDIM * 4);
  float*          bnsums = (float*)alloc((size_t)2 * 2048 * 4);
  float*          xls    = (float*)alloc((size_t)NNODES * 4);
  float*          xrs    = (float*)alloc((size_t)NNODES * 4);
  float*          outsc  = (float*)alloc((size_t)NNODES * 4);
  float*          sc     = (float*)alloc((size_t)NNODES * 4);
  float*          keepv  = (float*)alloc((size_t)NNODES * 4);
  float*          g0     = (float*)alloc((size_t)NGRAPH * 2048 * 4);
  unsigned short* g0b    = (unsigned short*)alloc((size_t)NGRAPH * 2048 * 2);
  float*          g1     = (float*)alloc((size_t)NGRAPH * 256 * 4);
  unsigned short* g1b    = (unsigned short*)alloc((size_t)NGRAPH * 256 * 2);
  float*          g2     = (float*)alloc((size_t)NGRAPH * 128 * 4);
  unsigned short* g2b    = (unsigned short*)alloc((size_t)NGRAPH * 128 * 2);
  float*          g3     = (float*)alloc((size_t)NGRAPH * 64 * 4);

  static const int convBase[5] = {3, 12, 21, 30, 39};
  static const int fiA[5]  = {4, 128, 256, 512, 1024};   // layer-1 K padded to 32
  static const int hcA[5]  = {128, 256, 512, 1024, 2048};
  unsigned short* wlB[5];
  unsigned short* wrB[5];
  for (int L = 0; L < 5; ++L) {
    int kp = (L == 0) ? 32 : fiA[L];
    wlB[L] = (unsigned short*)alloc((size_t)kp * hcA[L] * 2);
    wrB[L] = (unsigned short*)alloc((size_t)kp * hcA[L] * 2);
  }
  unsigned short* fc1b = (unsigned short*)alloc((size_t)2048 * 256 * 2);
  unsigned short* fc2b = (unsigned short*)alloc((size_t)256 * 128 * 2);
  unsigned short* fc3b = (unsigned short*)alloc((size_t)128 * 64 * 2);

  // ---- weight conversion (bf16); layer-1 zero-padded to K=32 ----
  hipMemsetAsync(wlB[0], 0, (size_t)32 * 128 * 2, stream);
  hipMemsetAsync(wrB[0], 0, (size_t)32 * 128 * 2, stream);
  cvt_bf16<<<(4 * 128 + 255) / 256, 256, 0, stream>>>(P(convBase[0] + 0), wlB[0], 4 * 128);
  cvt_bf16<<<(4 * 128 + 255) / 256, 256, 0, stream>>>(P(convBase[0] + 2), wrB[0], 4 * 128);
  for (int L = 1; L < 5; ++L) {
    int n = fiA[L] * hcA[L];
    cvt_bf16<<<(n + 255) / 256, 256, 0, stream>>>(P(convBase[L] + 0), wlB[L], n);
    cvt_bf16<<<(n + 255) / 256, 256, 0, stream>>>(P(convBase[L] + 2), wrB[L], n);
  }
  cvt_bf16<<<(2048 * 256 + 255) / 256, 256, 0, stream>>>(P(54), fc1b, 2048 * 256);
  cvt_bf16<<<(256 * 128 + 255) / 256, 256, 0, stream>>>(P(58), fc2b, 256 * 128);
  cvt_bf16<<<(128 * 64 + 255) / 256, 256, 0, stream>>>(P(62), fc3b, 128 * 64);

  // ---- self-loop edge attributes ----
  hipMemsetAsync(cnt, 0, (size_t)NNODES * 4, stream);
  hipMemsetAsync(lattr, 0, (size_t)NNODES * EDIM * 4, stream);
  edge_count_sum<<<(NEDGES + 255) / 256, 256, 0, stream>>>(dstA, ef, cnt, lattr, NEDGES);
  loop_attr_fin<<<(NNODES * EDIM + 255) / 256, 256, 0, stream>>>(lattr, cnt, NNODES * EDIM);

  // ---- conv layers ----
  const float* hin = x;
  for (int L = 0; L < 5; ++L) {
    int HC = hcA[L], base = convBase[L];
    int Kg = (L == 0) ? 32 : fiA[L];     // GEMM K (layer-1 padded)
    int logC = __builtin_ctz(HC / 4);
    if (L == 0) {
      hipMemsetAsync(hb, 0, (size_t)NNODES * 32 * 2, stream);
      cvt_bf16_pad<<<(NNODES * 4 + 255) / 256, 256, 0, stream>>>(hin, hb, NNODES * 4, 4, 32);
    } else {
      int nconv = NNODES * fiA[L];
      cvt_bf16<<<(nconv + 255) / 256, 256, 0, stream>>>(hin, hb, nconv);
    }
    dim3 gg(HC / 64, NNODES / 128);
    gemm_bf16_wmma<<<gg, 256, 0, stream>>>(hb, wlB[L], P(base + 1), XL, NNODES, HC, Kg);
    gemm_bf16_wmma<<<gg, 256, 0, stream>>>(hb, wrB[L], P(base + 3), XR, NNODES, HC, Kg);
    hipMemsetAsync(h, 0, (size_t)NNODES * HC * 4, stream);
    hipMemsetAsync(menc, 0, (size_t)NNODES * 4 * 4, stream);
    hipMemsetAsync(den, 0, (size_t)NNODES * 4 * 4, stream);
    edge_alpha_kernel<<<NE2, 256, 0, stream>>>(XL, XR, P(base + 4), ef, lattr, P(base + 5),
                                               srcA, dstA, alpha, HC, logC, NEDGES, NE2);
    seg_max<<<(NE2 * 4 + 255) / 256, 256, 0, stream>>>(alpha, dstA, menc, NEDGES, NE2, 4);
    seg_expsum<<<(NE2 * 4 + 255) / 256, 256, 0, stream>>>(alpha, dstA, menc, den, NEDGES, NE2, 4);
    aggregate<<<NE2, 256, 0, stream>>>(XL, alpha, den, srcA, dstA, h, HC, logC, NEDGES, NE2);
    hipMemsetAsync(bnsums, 0, (size_t)2 * HC * 4, stream);
    dim3 gs((HC + 255) / 256, NNODES / 256);
    bn_stats<<<gs, 256, 0, stream>>>(h, bnsums, NNODES, HC, 256);
    size_t tot = (size_t)NNODES * HC;
    bn_elu<<<(unsigned)((tot + 255) / 256), 256, 0, stream>>>(h, bnsums, P(base + 7), P(base + 8), h, NNODES, HC);
    hin = h;
  }

  // ---- SAGPool score (GATv2 H=1,C=1): score params at idx 48..53 ----
  score_gemv<<<NNODES, 256, 0, stream>>>(h, P(48), P(50), P(49), P(51), xls, xrs, 2048);
  hipMemsetAsync(menc, 0, (size_t)NNODES * 4, stream);
  hipMemsetAsync(den, 0, (size_t)NNODES * 4, stream);
  hipMemsetAsync(outsc, 0, (size_t)NNODES * 4, stream);
  score_edge<<<(NE2 + 255) / 256, 256, 0, stream>>>(xls, xrs, P(52), srcA, dstA, alpha, NEDGES, NE2);
  seg_max<<<(NE2 + 255) / 256, 256, 0, stream>>>(alpha, dstA, menc, NEDGES, NE2, 1);
  seg_expsum<<<(NE2 + 255) / 256, 256, 0, stream>>>(alpha, dstA, menc, den, NEDGES, NE2, 1);
  score_agg<<<(NE2 + 255) / 256, 256, 0, stream>>>(xls, alpha, den, srcA, dstA, outsc, NEDGES, NE2);
  score_tanh<<<(NNODES + 255) / 256, 256, 0, stream>>>(outsc, P(53), sc, NNODES);
  topk_mask<<<NGRAPH, NPER, 0, stream>>>(sc, keepv);
  pool_mean<<<dim3(2048 / 256, NGRAPH), 256, 0, stream>>>(h, keepv, g0, 2048);

  // ---- MLP head (WMMA for fc1..fc3) ----
  cvt_bf16<<<(NGRAPH * 2048 + 255) / 256, 256, 0, stream>>>(g0, g0b, NGRAPH * 2048);
  gemm_bf16_wmma<<<dim3(256 / 64, 1), 256, 0, stream>>>(g0b, fc1b, P(55), g1, NGRAPH, 256, 2048);
  hipMemsetAsync(bnsums, 0, (size_t)2 * 256 * 4, stream);
  bn_stats<<<dim3(1, 1), 256, 0, stream>>>(g1, bnsums, NGRAPH, 256, NGRAPH);
  bn_elu<<<(NGRAPH * 256 + 255) / 256, 256, 0, stream>>>(g1, bnsums, P(56), P(57), g1, NGRAPH, 256);

  cvt_bf16<<<(NGRAPH * 256 + 255) / 256, 256, 0, stream>>>(g1, g1b, NGRAPH * 256);
  gemm_bf16_wmma<<<dim3(128 / 64, 1), 256, 0, stream>>>(g1b, fc2b, P(59), g2, NGRAPH, 128, 256);
  hipMemsetAsync(bnsums, 0, (size_t)2 * 128 * 4, stream);
  bn_stats<<<dim3(1, 1), 256, 0, stream>>>(g2, bnsums, NGRAPH, 128, NGRAPH);
  bn_elu<<<(NGRAPH * 128 + 255) / 256, 256, 0, stream>>>(g2, bnsums, P(60), P(61), g2, NGRAPH, 128);

  cvt_bf16<<<(NGRAPH * 128 + 255) / 256, 256, 0, stream>>>(g2, g2b, NGRAPH * 128);
  gemm_bf16_wmma<<<dim3(1, 1), 256, 0, stream>>>(g2b, fc3b, P(63), g3, NGRAPH, 64, 128);
  hipMemsetAsync(bnsums, 0, (size_t)2 * 64 * 4, stream);
  bn_stats<<<dim3(1, 1), 256, 0, stream>>>(g3, bnsums, NGRAPH, 64, NGRAPH);
  bn_elu<<<(NGRAPH * 64 + 255) / 256, 256, 0, stream>>>(g3, bnsums, P(64), P(65), g3, NGRAPH, 64);

  fc4_logsoftmax<<<NGRAPH, 64, 0, stream>>>(g3, P(66), P(67), (float*)d_out);
}
